// RGLRULayer_62792421868022
// MI455X (gfx1250) — matrix operand order repaired and checked by reading, not compiled, not run on hardware
//
#include <hip/hip_runtime.h>

#define BATCH   4
#define SEQT    8192
#define DIM     1024
#define MROWS   (BATCH * SEQT)        // 32768
#define NCHUNK  32
#define CHLEN   (SEQT / NCHUNK)       // 256
#define KSTEPS  (DIM / 32)            // 32

typedef __attribute__((ext_vector_type(16))) __bf16       v16bf;
typedef __attribute__((ext_vector_type(8)))  float        v8f;
typedef __attribute__((ext_vector_type(4)))  unsigned int u32x4;
typedef __attribute__((ext_vector_type(8)))  int          i32x8;
typedef __attribute__((ext_vector_type(4)))  int          i32x4;

// ---------- helpers ----------

static __device__ __forceinline__ unsigned short f2bf(float f) {
  unsigned u = __builtin_bit_cast(unsigned, f);
  return (unsigned short)((u + 0x7FFFu + ((u >> 16) & 1u)) >> 16);
}

static __device__ __forceinline__ float sigmoidf_(float x) {
  return 1.0f / (1.0f + __expf(-x));
}

static __device__ __forceinline__ float alpha_clamped(float log_decay) {
  float a = sigmoidf_(log_decay);
  return fminf(fmaxf(a, 1e-6f), 1.0f - 1e-6f);
}

// Pack a CDNA5 TDM D# for a 2D tile (data_size = 2 bytes) and issue the load.
// lds_off: byte offset of destination within workgroup LDS.
// gptr:    global address of tile start. row_stride in elements.
static __device__ __forceinline__ void tdm_load_tile_2d(
    unsigned lds_off, const void* gptr,
    unsigned tile_w, unsigned tile_h, unsigned row_stride_elems) {
  unsigned long long ga = (unsigned long long)(size_t)gptr;
  u32x4 g0;
  g0[0] = 1u;                                              // count=1, user mode
  g0[1] = lds_off;                                         // lds_addr
  g0[2] = (unsigned)(ga & 0xFFFFFFFFull);                  // global_addr[31:0]
  g0[3] = (unsigned)((ga >> 32) & 0x01FFFFFFull)           // global_addr[56:32]
        | (2u << 30);                                      // type=2 (image)
  i32x8 g1;
  g1[0] = (int)(1u << 16);                                 // data_size=1 -> 2 bytes
  g1[1] = (int)((row_stride_elems & 0xFFFFu) << 16);       // tensor_dim0[15:0]
  g1[2] = (int)((row_stride_elems >> 16)                   // tensor_dim0[31:16]
        | ((tile_h & 0xFFFFu) << 16));                     // tensor_dim1[15:0]
  g1[3] = (int)((tile_h >> 16) | (tile_w << 16));          // dim1 hi | tile_dim0
  g1[4] = (int)(tile_h & 0xFFFFu);                         // tile_dim1 (tile_dim2=0)
  g1[5] = (int)row_stride_elems;                           // tensor_dim0_stride[31:0]
  g1[6] = 0;
  g1[7] = 0;
  i32x4 gz4; gz4[0] = 0; gz4[1] = 0; gz4[2] = 0; gz4[3] = 0;
  i32x8 gz8;
#pragma unroll
  for (int i = 0; i < 8; ++i) gz8[i] = 0;
  // 6-arg form (clang-23 / therock-10.0 headers)
  __builtin_amdgcn_tensor_load_to_lds(g0, g1, gz4, gz4, gz8, 0);
}

struct BfPair { uint4 a, b; };
static __device__ __forceinline__ v16bf load_frag(const unsigned char* base,
                                                  unsigned o0, unsigned o1) {
  BfPair s;
  s.a = *(const uint4*)(base + o0);
  s.b = *(const uint4*)(base + o1);
  return __builtin_bit_cast(v16bf, s);
}

// ---------- fp32 -> bf16 convert ----------

__global__ __launch_bounds__(256) void f2bf_kernel(const float* __restrict__ src,
                                                   unsigned short* __restrict__ dst,
                                                   int n) {
  int i = blockIdx.x * 256 + threadIdx.x;
  if (i < n) dst[i] = f2bf(src[i]);
}

// ---------- fused 3-way projection GEMM ----------
// X (MROWS x DIM, bf16) times W_in^T / W_r^T / W_i^T (each DIM x DIM, row-major N x K).
// Block: 256 thr (8 waves). Tile: M=128, N=64. Wave w owns rows [w*16, w*16+16).
// Inner loop: issue ALL ds_load_b128s, one s_wait_dscnt, then 12 back-to-back WMMAs.
// Epilogue: r=sig(.+b_r), i=sig(.+b_i), u=sqrt(1-a^2)*r*xproj -> U, Ig (fp32).

__global__ __launch_bounds__(256)
void rglru_gemm3_kernel(const unsigned short* __restrict__ xb,
                        const unsigned short* __restrict__ wrb,
                        const unsigned short* __restrict__ wib,
                        const unsigned short* __restrict__ winb,
                        const float* __restrict__ b_r,
                        const float* __restrict__ b_i,
                        const float* __restrict__ log_decay,
                        float* __restrict__ U,
                        float* __restrict__ Ig) {
  __shared__ __align__(16) unsigned char smem[40960];   // A: 2x8K, B: 3 mats x 2x4K
  const int lane  = threadIdx.x & 31;
  const int wave  = threadIdx.x >> 5;
  const int row16 = lane & 15;
  const int khalf = lane >> 4;
  const int nBase  = blockIdx.x * 64;
  const int mBlock = blockIdx.y * 128;
  const unsigned smemBase = (unsigned)(size_t)(&smem[0]);
  const unsigned ldsA = 0, ldsB = 16384;

  v8f accX[4], accR[4], accI[4];
#pragma unroll
  for (int nt = 0; nt < 4; ++nt)
#pragma unroll
    for (int j = 0; j < 8; ++j) { accX[nt][j] = 0.f; accR[nt][j] = 0.f; accI[nt][j] = 0.f; }

  if (wave == 0) {  // prologue: stage 0
    tdm_load_tile_2d(smemBase + ldsA,               xb  + (size_t)mBlock * DIM, 32, 128, DIM);
    tdm_load_tile_2d(smemBase + ldsB + 0u * 8192u,  winb + (size_t)nBase * DIM, 32, 64, DIM);
    tdm_load_tile_2d(smemBase + ldsB + 1u * 8192u,  wrb  + (size_t)nBase * DIM, 32, 64, DIM);
    tdm_load_tile_2d(smemBase + ldsB + 2u * 8192u,  wib  + (size_t)nBase * DIM, 32, 64, DIM);
  }

  for (int kk = 0; kk < KSTEPS; ++kk) {
    if (wave == 0) {
      if (kk + 1 < KSTEPS) {
        const unsigned s = (unsigned)(kk + 1) & 1u;
        const int k0 = (kk + 1) * 32;
        tdm_load_tile_2d(smemBase + ldsA + s * 8192u,              xb  + (size_t)mBlock * DIM + k0, 32, 128, DIM);
        tdm_load_tile_2d(smemBase + ldsB + 0u * 8192u + s * 4096u, winb + (size_t)nBase * DIM + k0, 32, 64, DIM);
        tdm_load_tile_2d(smemBase + ldsB + 1u * 8192u + s * 4096u, wrb  + (size_t)nBase * DIM + k0, 32, 64, DIM);
        tdm_load_tile_2d(smemBase + ldsB + 2u * 8192u + s * 4096u, wib  + (size_t)nBase * DIM + k0, 32, 64, DIM);
        __builtin_amdgcn_s_wait_tensorcnt(4);   // previous stage complete
      } else {
        __builtin_amdgcn_s_wait_tensorcnt(0);
      }
    }
    __syncthreads();

    const unsigned s = (unsigned)kk & 1u;
    const unsigned char* aP = smem + ldsA + s * 8192u;
    const unsigned arow = (unsigned)(wave * 16 + row16);
    // A frag: two 16B chunks per ISA 16-bit A layout
    v16bf afrag = load_frag(aP, arow * 64u + (unsigned)khalf * 16u,
                                arow * 64u + (unsigned)khalf * 16u + 32u);

    // load ALL B fragments first so the WMMA chain runs without dscnt stalls
    v16bf fX[4], fR[4], fI[4];
#pragma unroll
    for (int nt = 0; nt < 4; ++nt) {
      const unsigned boff = (unsigned)(nt * 16 + row16) * 64u + (unsigned)khalf * 16u;
      const unsigned char* bx = smem + ldsB + 0u * 8192u + s * 4096u;
      const unsigned char* br = smem + ldsB + 1u * 8192u + s * 4096u;
      const unsigned char* bi = smem + ldsB + 2u * 8192u + s * 4096u;
      fX[nt] = load_frag(bx, boff, boff + 32u);
      fR[nt] = load_frag(br, boff, boff + 32u);
      fI[nt] = load_frag(bi, boff, boff + 32u);
    }
#pragma unroll
    for (int nt = 0; nt < 4; ++nt) {
      accX[nt] = __builtin_amdgcn_wmma_f32_16x16x32_bf16(false, afrag, false, fX[nt],
                                                         (short)0, accX[nt], false, false);
      accR[nt] = __builtin_amdgcn_wmma_f32_16x16x32_bf16(false, afrag, false, fR[nt],
                                                         (short)0, accR[nt], false, false);
      accI[nt] = __builtin_amdgcn_wmma_f32_16x16x32_bf16(false, afrag, false, fI[nt],
                                                         (short)0, accI[nt], false, false);
    }
    __syncthreads();
  }

  // epilogue: gates + variance-preserving scale
#pragma unroll
  for (int nt = 0; nt < 4; ++nt) {
    const int col = nBase + nt * 16 + row16;
    const float a     = alpha_clamped(log_decay[col]);
    const float scale = sqrtf(fmaxf(1.0f - a * a, 0.0f));
    const float br = b_r[col];
    const float bi = b_i[col];
#pragma unroll
    for (int j = 0; j < 8; ++j) {
      const int m = mBlock + wave * 16 + khalf * 8 + j;  // C layout: vgpr j, lane half
      const float rv = sigmoidf_(accR[nt][j] + br);
      const float iv = sigmoidf_(accI[nt][j] + bi);
      const size_t o = (size_t)m * DIM + col;
      U[o]  = scale * rv * accX[nt][j];
      Ig[o] = iv;
    }
  }
}

// ---------- scan pass 1: local prefix scan (h0 = 0), in place over U ----------

__global__ __launch_bounds__(256)
void scan_local_kernel(const float* __restrict__ log_decay,
                       float* __restrict__ U,
                       float* __restrict__ carry) {
  const int d_ = blockIdx.x * 256 + threadIdx.x;
  const int c  = blockIdx.y;
  const int b  = blockIdx.z;
  const float a = alpha_clamped(log_decay[d_]);
  float h = 0.0f;
  size_t base = ((size_t)b * SEQT + (size_t)c * CHLEN) * DIM + d_;
  for (int t = 0; t < CHLEN; ++t) {
    h = fmaf(a, h, U[base]);
    U[base] = h;              // l_t
    base += DIM;
  }
  carry[((size_t)b * NCHUNK + c) * DIM + d_] = h;   // l_{CHLEN-1}
}

// ---------- scan pass 2: sequential carry combine across 32 chunks ----------

__global__ __launch_bounds__(256)
void scan_carry_kernel(const float* __restrict__ log_decay,
                       const float* __restrict__ state,
                       const float* __restrict__ carry,
                       float* __restrict__ Hc,
                       float* __restrict__ h_final) {
  const int idx = blockIdx.x * 256 + threadIdx.x;   // b*DIM + d
  const int b = idx >> 10, d_ = idx & (DIM - 1);
  const float a  = alpha_clamped(log_decay[d_]);
  const float aL = __expf((float)CHLEN * __logf(a));  // alpha^CHLEN
  float H = state[idx];                                // carry INTO chunk 0
  for (int c = 0; c < NCHUNK; ++c) {
    const size_t o = ((size_t)b * NCHUNK + c) * DIM + d_;
    Hc[o] = H;
    H = fmaf(aL, H, carry[o]);
  }
  h_final[idx] = H;
}

// ---------- scan pass 3: h_t = l_t + a^{t+1} H_in ; y = h * i -> bf16 ----------

__global__ __launch_bounds__(256)
void scan_fixup_kernel(const float* __restrict__ log_decay,
                       const float* __restrict__ U,   // l values
                       const float* __restrict__ Ig,
                       const float* __restrict__ Hc,
                       unsigned short* __restrict__ ysb) {
  const int d_ = blockIdx.x * 256 + threadIdx.x;
  const int c  = blockIdx.y;
  const int b  = blockIdx.z;
  const float a = alpha_clamped(log_decay[d_]);
  const float H = Hc[((size_t)b * NCHUNK + c) * DIM + d_];
  float ap = a;                                     // alpha^{t+1}
  size_t base = ((size_t)b * SEQT + (size_t)c * CHLEN) * DIM + d_;
  for (int t = 0; t < CHLEN; ++t) {
    const float h = fmaf(ap, H, U[base]);
    ysb[base] = f2bf(h * Ig[base]);
    ap *= a;
    base += DIM;
  }
}

// ---------- output GEMM: ys (bf16) @ W_out^T -> out (fp32) ----------

__global__ __launch_bounds__(256)
void rglru_gemm_out_kernel(const unsigned short* __restrict__ ysb,
                           const unsigned short* __restrict__ woutb,
                           float* __restrict__ out) {
  __shared__ __align__(16) unsigned char smem[24576];  // A: 2x8K, B: 2x4K
  const int lane  = threadIdx.x & 31;
  const int wave  = threadIdx.x >> 5;
  const int row16 = lane & 15;
  const int khalf = lane >> 4;
  const int nBase  = blockIdx.x * 64;
  const int mBlock = blockIdx.y * 128;
  const unsigned smemBase = (unsigned)(size_t)(&smem[0]);
  const unsigned ldsA = 0, ldsB = 16384;

  v8f acc[4];
#pragma unroll
  for (int nt = 0; nt < 4; ++nt)
#pragma unroll
    for (int j = 0; j < 8; ++j) acc[nt][j] = 0.f;

  if (wave == 0) {
    tdm_load_tile_2d(smemBase + ldsA, ysb   + (size_t)mBlock * DIM, 32, 128, DIM);
    tdm_load_tile_2d(smemBase + ldsB, woutb + (size_t)nBase * DIM, 32, 64, DIM);
  }

  for (int kk = 0; kk < KSTEPS; ++kk) {
    if (wave == 0) {
      if (kk + 1 < KSTEPS) {
        const unsigned s = (unsigned)(kk + 1) & 1u;
        const int k0 = (kk + 1) * 32;
        tdm_load_tile_2d(smemBase + ldsA + s * 8192u, ysb   + (size_t)mBlock * DIM + k0, 32, 128, DIM);
        tdm_load_tile_2d(smemBase + ldsB + s * 4096u, woutb + (size_t)nBase * DIM + k0, 32, 64, DIM);
        __builtin_amdgcn_s_wait_tensorcnt(2);
      } else {
        __builtin_amdgcn_s_wait_tensorcnt(0);
      }
    }
    __syncthreads();

    const unsigned s = (unsigned)kk & 1u;
    const unsigned char* aP = smem + ldsA + s * 8192u;
    const unsigned char* bP = smem + ldsB + s * 4096u;
    const unsigned arow = (unsigned)(wave * 16 + row16);
    v16bf afrag = load_frag(aP, arow * 64u + (unsigned)khalf * 16u,
                                arow * 64u + (unsigned)khalf * 16u + 32u);
    v16bf fB[4];
#pragma unroll
    for (int nt = 0; nt < 4; ++nt) {
      const unsigned boff = (unsigned)(nt * 16 + row16) * 64u + (unsigned)khalf * 16u;
      fB[nt] = load_frag(bP, boff, boff + 32u);
    }
#pragma unroll
    for (int nt = 0; nt < 4; ++nt) {
      acc[nt] = __builtin_amdgcn_wmma_f32_16x16x32_bf16(false, afrag, false, fB[nt],
                                                        (short)0, acc[nt], false, false);
    }
    __syncthreads();
  }

#pragma unroll
  for (int nt = 0; nt < 4; ++nt) {
    const int col = nBase + nt * 16 + row16;
#pragma unroll
    for (int j = 0; j < 8; ++j) {
      const int m = mBlock + wave * 16 + khalf * 8 + j;
      out[(size_t)m * DIM + col] = acc[nt][j];
    }
  }
}

// ---------- launch ----------

// workspace byte offsets
#define OFF_XB    0ull                       // 32768*1024 bf16  = 64 MB
#define OFF_WR    67108864ull                // 1024*1024 bf16   = 2 MB
#define OFF_WI    69206016ull
#define OFF_WIN   71303168ull
#define OFF_WOUT  73400320ull
#define OFF_U     75497472ull                // 32768*1024 fp32  = 128 MB
#define OFF_I     209715200ull
#define OFF_YS    343932928ull               // bf16, 64 MB
#define OFF_CARRY 411041792ull               // 4*32*1024 fp32   = 512 KB
#define OFF_HC    411566080ull

extern "C" void kernel_launch(void* const* d_in, const int* in_sizes, int n_in,
                              void* d_out, int out_size, void* d_ws, size_t ws_size,
                              hipStream_t stream) {
  (void)in_sizes; (void)n_in; (void)out_size; (void)ws_size;
  const float* x         = (const float*)d_in[0];
  const float* state     = (const float*)d_in[1];
  const float* W_in      = (const float*)d_in[2];
  const float* W_r       = (const float*)d_in[3];
  const float* b_r       = (const float*)d_in[4];
  const float* W_i       = (const float*)d_in[5];
  const float* b_i       = (const float*)d_in[6];
  const float* log_decay = (const float*)d_in[7];
  const float* W_out     = (const float*)d_in[8];

  unsigned char* ws = (unsigned char*)d_ws;
  unsigned short* xb    = (unsigned short*)(ws + OFF_XB);
  unsigned short* wrb   = (unsigned short*)(ws + OFF_WR);
  unsigned short* wib   = (unsigned short*)(ws + OFF_WI);
  unsigned short* winb  = (unsigned short*)(ws + OFF_WIN);
  unsigned short* woutb = (unsigned short*)(ws + OFF_WOUT);
  float*          U     = (float*)(ws + OFF_U);
  float*          Ig    = (float*)(ws + OFF_I);
  unsigned short* ysb   = (unsigned short*)(ws + OFF_YS);
  float*          carry = (float*)(ws + OFF_CARRY);
  float*          Hc    = (float*)(ws + OFF_HC);

  float* out     = (float*)d_out;                       // (B,T,d) fp32
  float* h_final = out + (size_t)MROWS * DIM;           // (B,d) fp32

  const int NX = MROWS * DIM;   // 33554432
  const int NW = DIM * DIM;     // 1048576

  f2bf_kernel<<<(NX + 255) / 256, 256, 0, stream>>>(x, xb, NX);
  f2bf_kernel<<<(NW + 255) / 256, 256, 0, stream>>>(W_r, wrb, NW);
  f2bf_kernel<<<(NW + 255) / 256, 256, 0, stream>>>(W_i, wib, NW);
  f2bf_kernel<<<(NW + 255) / 256, 256, 0, stream>>>(W_in, winb, NW);
  f2bf_kernel<<<(NW + 255) / 256, 256, 0, stream>>>(W_out, woutb, NW);

  rglru_gemm3_kernel<<<dim3(DIM / 64, MROWS / 128), 256, 0, stream>>>(
      xb, wrb, wib, winb, b_r, b_i, log_decay, U, Ig);

  scan_local_kernel<<<dim3(DIM / 256, NCHUNK, BATCH), 256, 0, stream>>>(log_decay, U, carry);
  scan_carry_kernel<<<(BATCH * DIM) / 256, 256, 0, stream>>>(log_decay, state, carry, Hc, h_final);
  scan_fixup_kernel<<<dim3(DIM / 256, NCHUNK, BATCH), 256, 0, stream>>>(log_decay, U, Ig, Hc, ysb);

  rglru_gemm_out_kernel<<<dim3(DIM / 64, MROWS / 128), 256, 0, stream>>>(ysb, woutb, out);
}
